// ClusterMemory_26560077758538
// MI455X (gfx1250) — matrix operand order repaired and checked by reading, not compiled, not run on hardware
//
#include <hip/hip_runtime.h>
#include <hip/hip_bf16.h>

typedef __attribute__((ext_vector_type(16))) _Float16 v16h;
typedef __attribute__((ext_vector_type(8)))  float    v8f;

#define B_SZ   512
#define D_SZ   1024
#define N_SZ   16384
#define SCALE  20.0f      // 1/TEMP
#define MAXL   20.0f      // |cosine|/TEMP <= 20, fixed logsumexp shift
#define CHUNKS 256        // 64 cols per chunk

// ---------------- K1: L2-normalize inputs into workspace ----------------
__global__ void k_normalize(const float* __restrict__ inR,
                            const float* __restrict__ inI,
                            float* __restrict__ xn) {
    const int mod = blockIdx.x >> 9;
    const int row = blockIdx.x & 511;
    const float* src = (mod ? inI : inR) + (size_t)row * D_SZ;
    float* dst = xn + ((size_t)mod * B_SZ + row) * D_SZ;
    __shared__ float sd[256];
    float s = 0.f;
    for (int j = threadIdx.x; j < D_SZ; j += 256) { float v = src[j]; s += v * v; }
    sd[threadIdx.x] = s; __syncthreads();
    for (int off = 128; off > 0; off >>= 1) {
        if (threadIdx.x < off) sd[threadIdx.x] += sd[threadIdx.x + off];
        __syncthreads();
    }
    const float inv = 1.0f / fmaxf(sqrtf(sd[0]), 1e-12f);
    for (int j = threadIdx.x; j < D_SZ; j += 256) dst[j] = src[j] * inv;
}

// ---------------- K2: WMMA GEMM + fused softmax partials ----------------
// grid (256 col-chunks, 32 row-strips, 2 modalities), block = 1 wave (32).
// Each wave: 16 rows x 64 cols, K=1024 -> 32 k-steps x 4 WMMAs.
__global__ void k_logits(const float* __restrict__ xn,
                         const float* __restrict__ fR,
                         const float* __restrict__ fI,
                         const int* __restrict__ tR,
                         const int* __restrict__ tI,
                         float* __restrict__ partial,
                         float* __restrict__ tgtlog) {
    const int lane    = threadIdx.x;
    const int chunk   = blockIdx.x;          // 64 cols
    const int rs      = blockIdx.y;          // 16 rows
    const int mod     = blockIdx.z;
    const float* F    = mod ? fI : fR;
    const int*   T    = mod ? tI : tR;
    const int rowBase = rs * 16;
    const int colBase = chunk * 64;
    const int aRow    = lane & 15;
    const int hi      = lane >> 4;           // 0 or 1 (lane half)

    const float* aPtr = xn + ((size_t)(mod * B_SZ + rowBase + aRow)) * D_SZ + hi * 16;

    v8f c0 = {}, c1 = {}, c2 = {}, c3 = {};

    for (int k = 0; k < D_SZ; k += 32) {
        // A tile 16x32 f16: lane holds row aRow, K = k + hi*16 + 0..15
        v16h a;
        const float* p = aPtr + k;
#pragma unroll
        for (int j = 0; j < 8; j++) {
            a[j]     = (_Float16)p[j];
            a[8 + j] = (_Float16)p[8 + j];
        }
        // B tile 32x16 f16 per col-tile: lane holds column n, K = k + hi*16 + 0..15
#define WSTEP(CC, t)                                                          \
        {                                                                     \
            const float* q = F + ((size_t)(colBase + (t) * 16 + aRow)) * D_SZ \
                               + k + hi * 16;                                 \
            v16h b;                                                           \
            _Pragma("unroll")                                                 \
            for (int j = 0; j < 16; j++) b[j] = (_Float16)q[j];               \
            CC = __builtin_amdgcn_wmma_f32_16x16x32_f16(                      \
                false, a, false, b, (short)0, CC, false, false);              \
        }
        WSTEP(c0, 0) WSTEP(c1, 1) WSTEP(c2, 2) WSTEP(c3, 3)
#undef WSTEP
    }

    // logits = dot * (1/TEMP)
    c0 *= SCALE; c1 *= SCALE; c2 *= SCALE; c3 *= SCALE;

    // Row partial sums of exp(l - MAXL): butterfly over the 16-lane halves.
    // VGPR i: lanes 0-15 -> row rowBase+i, lanes 16-31 -> row rowBase+8+i.
    float psum[8];
#pragma unroll
    for (int i = 0; i < 8; i++) {
        float e = __expf(c0[i] - MAXL) + __expf(c1[i] - MAXL)
                + __expf(c2[i] - MAXL) + __expf(c3[i] - MAXL);
        e += __shfl_xor(e, 1, 32);
        e += __shfl_xor(e, 2, 32);
        e += __shfl_xor(e, 4, 32);
        e += __shfl_xor(e, 8, 32);
        psum[i] = e;
    }
    if (lane == 0) {
#pragma unroll
        for (int i = 0; i < 8; i++)
            partial[((size_t)(mod * B_SZ + rowBase + i)) * CHUNKS + chunk] = psum[i];
    }
    if (lane == 16) {
#pragma unroll
        for (int i = 0; i < 8; i++)
            partial[((size_t)(mod * B_SZ + rowBase + 8 + i)) * CHUNKS + chunk] = psum[i];
    }

    // Target logit extraction: exactly one chunk owns each target column.
#pragma unroll
    for (int m = 0; m < 16; m++) {
        const int tg    = T[rowBase + m];
        const int local = tg - colBase;
        if (local >= 0 && local < 64) {
            const int t   = local >> 4;
            const int n   = local & 15;
            const int src = (m < 8) ? n : (n + 16);
            v8f pick = (t == 0) ? c0 : (t == 1) ? c1 : (t == 2) ? c2 : c3;
            const float v = __shfl(pick[m & 7], src, 32);
            if (lane == 0) tgtlog[mod * B_SZ + rowBase + m] = v;
        }
    }
}

// ---------------- K3: finalize cross-entropy losses ----------------
__global__ void k_loss(const float* __restrict__ partial,
                       const float* __restrict__ tgtlog,
                       float* __restrict__ out) {
    const int mod = blockIdx.x;
    __shared__ float sd[256];
    float acc = 0.f;
    for (int r = 0; r < B_SZ; r++) {
        sd[threadIdx.x] = partial[((size_t)(mod * B_SZ + r)) * CHUNKS + threadIdx.x];
        __syncthreads();
        for (int off = 128; off > 0; off >>= 1) {
            if (threadIdx.x < off) sd[threadIdx.x] += sd[threadIdx.x + off];
            __syncthreads();
        }
        if (threadIdx.x == 0)
            acc += logf(sd[0]) + MAXL - tgtlog[mod * B_SZ + r];
        __syncthreads();
    }
    if (threadIdx.x == 0) out[mod] = acc / (float)B_SZ;
}

// ---------------- K4: stream feature bank -> output ----------------
__global__ void k_copy(const float* __restrict__ src, float* __restrict__ dst,
                       size_t n2) {
    const float2* s = (const float2*)src;
    float2*       d = (float2*)dst;   // dst is 8B-aligned (d_out + 2 floats)
    for (size_t i = (size_t)blockIdx.x * blockDim.x + threadIdx.x;
         i < n2; i += (size_t)gridDim.x * blockDim.x)
        d[i] = s[i];
}

// ---------------- K5: sequential momentum scatter-update ----------------
// One block per (modality, sample). Only the block owning the FIRST
// occurrence of its target row runs; it chains all duplicates in order.
__global__ void k_momentum(const float* __restrict__ xn,
                           const int* __restrict__ tR,
                           const int* __restrict__ tI,
                           const float* __restrict__ fR,
                           const float* __restrict__ fI,
                           float* __restrict__ outF) {
    const int mod = blockIdx.x >> 9;
    const int i   = blockIdx.x & 511;
    const int*   T = mod ? tI : tR;
    const float* F = mod ? fI : fR;
    const float* X = xn + (size_t)mod * B_SZ * D_SZ;
    float*       O = outF + (size_t)mod * N_SZ * D_SZ;
    const int y = T[i];

    __shared__ int notfirst;
    __shared__ float sd[256];
    if (threadIdx.x == 0) notfirst = 0;
    __syncthreads();
    for (int j = threadIdx.x; j < i; j += 256)
        if (T[j] == y) notfirst = 1;
    __syncthreads();
    if (notfirst) return;

    float r[4];
    const float* frow = F + (size_t)y * D_SZ;
#pragma unroll
    for (int e = 0; e < 4; e++) r[e] = frow[threadIdx.x * 4 + e];

    for (int j = i; j < B_SZ; j++) {
        if (T[j] != y) continue;                 // uniform across block
        const float* xr = X + (size_t)j * D_SZ;
        float ss = 0.f;
#pragma unroll
        for (int e = 0; e < 4; e++) {
            r[e] = 0.9f * r[e] + 0.1f * xr[threadIdx.x * 4 + e];
            ss += r[e] * r[e];
        }
        sd[threadIdx.x] = ss; __syncthreads();
        for (int off = 128; off > 0; off >>= 1) {
            if (threadIdx.x < off) sd[threadIdx.x] += sd[threadIdx.x + off];
            __syncthreads();
        }
        const float inv = 1.0f / sqrtf(sd[0]);
        __syncthreads();
#pragma unroll
        for (int e = 0; e < 4; e++) r[e] *= inv;
    }

    float* orow = O + (size_t)y * D_SZ;
#pragma unroll
    for (int e = 0; e < 4; e++) orow[threadIdx.x * 4 + e] = r[e];
}

extern "C" void kernel_launch(void* const* d_in, const int* in_sizes, int n_in,
                              void* d_out, int out_size, void* d_ws, size_t ws_size,
                              hipStream_t stream) {
    const float* inR = (const float*)d_in[0];
    const float* inI = (const float*)d_in[1];
    const int*   tR  = (const int*)d_in[2];
    const int*   tI  = (const int*)d_in[3];
    const float* fR  = (const float*)d_in[4];
    const float* fI  = (const float*)d_in[5];
    float* out = (float*)d_out;

    float* ws      = (float*)d_ws;
    float* xn      = ws;                                   // 2*512*1024 f32
    float* partial = xn + (size_t)2 * B_SZ * D_SZ;         // 2*512*256  f32
    float* tgtlog  = partial + (size_t)2 * B_SZ * CHUNKS;  // 2*512      f32

    k_normalize<<<2 * B_SZ, 256, 0, stream>>>(inR, inI, xn);

    dim3 g2(CHUNKS, B_SZ / 16, 2);
    k_logits<<<g2, 32, 0, stream>>>(xn, fR, fI, tR, tI, partial, tgtlog);

    k_loss<<<2, 256, 0, stream>>>(partial, tgtlog, out);

    const size_t NF = (size_t)N_SZ * D_SZ;
    k_copy<<<4096, 256, 0, stream>>>(fR, out + 2, NF / 2);
    k_copy<<<4096, 256, 0, stream>>>(fI, out + 2 + NF, NF / 2);

    k_momentum<<<2 * B_SZ, 256, 0, stream>>>(xn, tR, tI, fR, fI, out + 2);
}
